// CausalSelfAttention_89292370084006
// MI455X (gfx1250) — matrix operand order repaired and checked by reading, not compiled, not run on hardware
//
#include <hip/hip_runtime.h>
#include <hip/hip_bf16.h>

// ---------- problem constants ----------
#define BB 4
#define TT 2048
#define CC 768
#define HH 12
#define DD 64
#define C3 2304

typedef _Float16 v16h __attribute__((ext_vector_type(16)));
typedef _Float16 v8h  __attribute__((ext_vector_type(8)));
typedef float    v8f  __attribute__((ext_vector_type(8)));
typedef int      v4i_vs __attribute__((vector_size(16)));   // b128 unit for async builtin

#define LDS_PITCH 40   // f16 per staged row: 32 data + 8 pad -> conflict-free ds reads
#define ABUF (64 * LDS_PITCH)
#define BBUF (128 * LDS_PITCH)

#if __has_builtin(__builtin_amdgcn_global_load_async_to_lds_b128)
#define USE_ASYNC_LDS 1
#else
#define USE_ASYNC_LDS 0
#endif

__device__ __forceinline__ v16h ld_frag(const _Float16* lo, const _Float16* hi) {
    v8h a = *(const v8h*)lo;
    v8h b = *(const v8h*)hi;
    return __builtin_shufflevector(a, b, 0,1,2,3,4,5,6,7,8,9,10,11,12,13,14,15);
}

__device__ __forceinline__ v8f wmma_f16(v16h a, v16h b, v8f c) {
    return __builtin_amdgcn_wmma_f32_16x16x32_f16(false, a, false, b, (short)0, c, false, false);
}

// 16-byte global -> LDS copy (async on gfx1250 toolchains that expose it)
__device__ __forceinline__ void copy16_g2l(const _Float16* g, _Float16* l) {
#if USE_ASYNC_LDS
    __builtin_amdgcn_global_load_async_to_lds_b128((v4i_vs*)g, (v4i_vs*)l, 0, 0);
#else
    *(v8h*)l = *(const v8h*)g;
#endif
}

__device__ __forceinline__ void g2l_drain() {
#if USE_ASYNC_LDS
#if __has_builtin(__builtin_amdgcn_s_wait_asynccnt)
    __builtin_amdgcn_s_wait_asynccnt(0);
#else
    asm volatile("s_wait_asynccnt 0x0" ::: "memory");
#endif
#endif
}

// ---------------------------------------------------------------
// Kernel 1: convert x -> f16, W_qkv^T -> f16, W_proj^T -> f16
// ---------------------------------------------------------------
#define NX  (BB*TT*CC)      // 6291456
#define NWQ (C3*CC)         // 1769472
#define NWP (CC*CC)         // 589824

__global__ __launch_bounds__(256)
void k_convert(const float* __restrict__ x, const float* __restrict__ wqkv,
               const float* __restrict__ wproj,
               _Float16* __restrict__ xh, _Float16* __restrict__ wqkvT,
               _Float16* __restrict__ wprojT) {
    size_t i = (size_t)blockIdx.x * blockDim.x + threadIdx.x;
    if (i < (size_t)NX) {
        xh[i] = (_Float16)x[i];
    } else if (i < (size_t)(NX + NWQ)) {
        size_t j = i - NX;
        int n = (int)(j / CC), k = (int)(j % CC);
        wqkvT[j] = (_Float16)wqkv[(size_t)k * C3 + n];
    } else if (i < (size_t)(NX + NWQ + NWP)) {
        size_t j = i - NX - NWQ;
        int n = (int)(j / CC), k = (int)(j % CC);
        wprojT[j] = (_Float16)wproj[(size_t)k * CC + n];
    }
}

// ---------------------------------------------------------------
// Shared GEMM core: 256 threads (8 waves, 4M x 2N), block tile 64x128,
// double-buffered LDS staging.  acc[4] = wave's 16x64 slab.
// ---------------------------------------------------------------
__device__ __forceinline__ void gemm64x128(const _Float16* __restrict__ A,
                                           const _Float16* __restrict__ Bt,
                                           int row0, int c0,
                                           _Float16* ldsA, _Float16* ldsB,
                                           v8f acc[4]) {
    const int tid  = threadIdx.x;
    const int lane = tid & 31;
    const int wave = tid >> 5;
    const int m    = lane & 15;
    const int half = lane >> 4;
    const int wm   = wave & 3;     // M sub-tile (16 rows each)
    const int wn   = wave >> 2;    // N half (64 cols each)

    const int arow = tid >> 2;              // 0..63
    const int aseg = (tid & 3) * 8;         // f16 offset within 32-K row

    auto stage = [&](int buf, int kk) {
        // A tile: 64 rows x 32 K  (one b128 per thread)
        copy16_g2l(A + (size_t)(row0 + arow) * CC + kk + aseg,
                   ldsA + buf * ABUF + arow * LDS_PITCH + aseg);
        // B tile: 128 cols x 32 K (two b128 per thread)
#pragma unroll
        for (int u = 0; u < 2; ++u) {
            const int idx = tid + u * 256;
            const int col = idx >> 2;
            const int seg = (idx & 3) * 8;
            copy16_g2l(Bt + (size_t)(c0 + col) * CC + kk + seg,
                       ldsB + buf * BBUF + col * LDS_PITCH + seg);
        }
    };

    stage(0, 0);
    int buf = 0;
    for (int kk = 0; kk < CC; kk += 32, buf ^= 1) {
        g2l_drain();          // own async copies for buf done
        __syncthreads();      // everyone's copies done; prior buf fully consumed
        if (kk + 32 < CC) stage(buf ^ 1, kk + 32);

        const _Float16* la = ldsA + buf * ABUF + (wm * 16 + m) * LDS_PITCH;
        v16h a = ld_frag(la + half * 8, la + 16 + half * 8);
#pragma unroll
        for (int t = 0; t < 4; ++t) {
            const _Float16* lb = ldsB + buf * BBUF +
                                 (wn * 64 + t * 16 + m) * LDS_PITCH + half * 8;
            v16h b = ld_frag(lb, lb + 16);
            acc[t] = wmma_f16(a, b, acc[t]);
        }
    }
}

// ---------------------------------------------------------------
// Kernel 2: QKV GEMM.  Q,K -> [B,H,T,D] f16, V -> [B,H,D,T] f16.
// ---------------------------------------------------------------
__global__ __launch_bounds__(256)
void k_qkv(const _Float16* __restrict__ xh, const _Float16* __restrict__ wt,
           const float* __restrict__ bias,
           _Float16* __restrict__ qb, _Float16* __restrict__ kbuf,
           _Float16* __restrict__ vbuf) {
    __shared__ _Float16 ldsA[2 * ABUF];
    __shared__ _Float16 ldsB[2 * BBUF];

    const int row0 = blockIdx.x * 64;     // 0..8128
    const int c0   = blockIdx.y * 128;    // 0..2176

    v8f acc[4] = {v8f{}, v8f{}, v8f{}, v8f{}};
    gemm64x128(xh, wt, row0, c0, ldsA, ldsB, acc);

    const int lane = threadIdx.x & 31;
    const int wave = threadIdx.x >> 5;
    const int m    = lane & 15;
    const int half = lane >> 4;
    const int wm   = wave & 3;
    const int wn   = wave >> 2;
    const int r0   = row0 + wm * 16;

#pragma unroll
    for (int t = 0; t < 4; ++t) {
        const int c     = c0 + wn * 64 + t * 16 + m;  // 0..2303
        const int which = c / CC;
        const int cc    = c % CC;
        const int h     = cc / DD;
        const int d     = cc % DD;
        const float bv  = bias[c];
#pragma unroll
        for (int r = 0; r < 8; ++r) {
            const int rg = r0 + r + 8 * half;   // b*T + t
            const int bi = rg >> 11;
            const int ti = rg & (TT - 1);
            _Float16 hv = (_Float16)(acc[t][r] + bv);
            if (which == 0)
                qb[(((size_t)bi * HH + h) * TT + ti) * DD + d] = hv;
            else if (which == 1)
                kbuf[(((size_t)bi * HH + h) * TT + ti) * DD + d] = hv;
            else
                vbuf[(((size_t)bi * HH + h) * DD + d) * TT + ti] = hv;
        }
    }
}

// ---------------------------------------------------------------
// Kernel 3: causal flash attention.  One wave per 16-query tile,
// 32 keys/step: 4 WMMA for S, online softmax, LDS C->A transpose,
// 4 WMMA for O += P*V.
// ---------------------------------------------------------------
__global__ __launch_bounds__(128)
void k_attn(const _Float16* __restrict__ q, const _Float16* __restrict__ k,
            const _Float16* __restrict__ vt, _Float16* __restrict__ y) {
    __shared__ _Float16 lds[4 * 16 * 32];

    const int lane = threadIdx.x & 31;
    const int wave = threadIdx.x >> 5;
    const int m    = lane & 15;
    const int half = lane >> 4;

    const int wid = blockIdx.x * 4 + wave;        // 0..6143
    const int b   = wid / (HH * 128);
    const int h   = (wid / 128) % HH;
    const int qt  = wid & 127;
    const int t0  = qt * 16;

    const _Float16* qrow  = q  + (((size_t)b * HH + h) * TT + (t0 + m)) * DD;
    const _Float16* kbase = k  + (((size_t)b * HH + h) * TT) * DD;
    const _Float16* vbase = vt + (((size_t)b * HH + h) * DD) * TT;
    _Float16* myl = lds + wave * (16 * 32);

    v16h aq0 = ld_frag(qrow + half * 8,      qrow + 16 + half * 8);
    v16h aq1 = ld_frag(qrow + 32 + half * 8, qrow + 48 + half * 8);

    float mrow[8], lrow[8];
#pragma unroll
    for (int r = 0; r < 8; ++r) { mrow[r] = -1e30f; lrow[r] = 0.0f; }
    v8f o[4] = {v8f{}, v8f{}, v8f{}, v8f{}};

    const float scale = 0.125f;  // 1/sqrt(64)

    for (int kb = 0; kb < t0 + 16; kb += 32) {
        v8f s0 = v8f{}, s1 = v8f{};
        {
            const _Float16* kp0 = kbase + (size_t)(kb + m) * DD + half * 8;
            v16h b0a = ld_frag(kp0,      kp0 + 16);
            v16h b0b = ld_frag(kp0 + 32, kp0 + 48);
            const _Float16* kp1 = kbase + (size_t)(kb + 16 + m) * DD + half * 8;
            v16h b1a = ld_frag(kp1,      kp1 + 16);
            v16h b1b = ld_frag(kp1 + 32, kp1 + 48);
            s0 = wmma_f16(aq0, b0a, s0);
            s0 = wmma_f16(aq1, b0b, s0);
            s1 = wmma_f16(aq0, b1a, s1);
            s1 = wmma_f16(aq1, b1b, s1);
        }

        const bool diag = (kb + 31 > t0);
        const int k0 = kb + m, k1 = kb + 16 + m;

#pragma unroll
        for (int r = 0; r < 8; ++r) {
            const int rowg = t0 + r + 8 * half;
            float x0 = s0[r] * scale;
            float x1 = s1[r] * scale;
            if (diag) {
                if (k0 > rowg) x0 = -1e30f;
                if (k1 > rowg) x1 = -1e30f;
            }
            float bm = fmaxf(x0, x1);
            bm = fmaxf(bm, __shfl_xor(bm, 1, 32));
            bm = fmaxf(bm, __shfl_xor(bm, 2, 32));
            bm = fmaxf(bm, __shfl_xor(bm, 4, 32));
            bm = fmaxf(bm, __shfl_xor(bm, 8, 32));
            const float mnew  = fmaxf(mrow[r], bm);
            const float alpha = __expf(mrow[r] - mnew);
            const float p0 = __expf(x0 - mnew);
            const float p1 = __expf(x1 - mnew);
            float rs = p0 + p1;
            rs += __shfl_xor(rs, 1, 32);
            rs += __shfl_xor(rs, 2, 32);
            rs += __shfl_xor(rs, 4, 32);
            rs += __shfl_xor(rs, 8, 32);
            lrow[r] = lrow[r] * alpha + rs;
            mrow[r] = mnew;
#pragma unroll
            for (int t = 0; t < 4; ++t) o[t][r] *= alpha;
            myl[(r + 8 * half) * 32 + m]      = (_Float16)p0;
            myl[(r + 8 * half) * 32 + 16 + m] = (_Float16)p1;
        }

        asm volatile("s_wait_dscnt 0" ::: "memory");

        v16h ap = ld_frag(myl + m * 32 + half * 8, myl + m * 32 + 16 + half * 8);

#pragma unroll
        for (int t = 0; t < 4; ++t) {
            const _Float16* vp = vbase + (size_t)(t * 16 + m) * TT + kb + half * 8;
            v16h bv = ld_frag(vp, vp + 16);
            o[t] = wmma_f16(ap, bv, o[t]);
        }
    }

#pragma unroll
    for (int r = 0; r < 8; ++r) {
        const float inv = 1.0f / lrow[r];
        const size_t row = ((size_t)b * TT + t0 + r + 8 * half) * CC + h * DD;
#pragma unroll
        for (int t = 0; t < 4; ++t)
            y[row + t * 16 + m] = (_Float16)(o[t][r] * inv);
    }
}

// ---------------------------------------------------------------
// Kernel 4: output projection -> f32 out (+bias)
// ---------------------------------------------------------------
__global__ __launch_bounds__(256)
void k_proj(const _Float16* __restrict__ yh, const _Float16* __restrict__ wt,
            const float* __restrict__ bias, float* __restrict__ out) {
    __shared__ _Float16 ldsA[2 * ABUF];
    __shared__ _Float16 ldsB[2 * BBUF];

    const int row0 = blockIdx.x * 64;
    const int c0   = blockIdx.y * 128;

    v8f acc[4] = {v8f{}, v8f{}, v8f{}, v8f{}};
    gemm64x128(yh, wt, row0, c0, ldsA, ldsB, acc);

    const int lane = threadIdx.x & 31;
    const int wave = threadIdx.x >> 5;
    const int m    = lane & 15;
    const int half = lane >> 4;
    const int wm   = wave & 3;
    const int wn   = wave >> 2;
    const int r0   = row0 + wm * 16;

#pragma unroll
    for (int t = 0; t < 4; ++t) {
        const int c    = c0 + wn * 64 + t * 16 + m;
        const float bv = bias[c];
#pragma unroll
        for (int r = 0; r < 8; ++r) {
            const int rg = r0 + r + 8 * half;
            out[(size_t)rg * CC + c] = acc[t][r] + bv;
        }
    }
}

// ---------------------------------------------------------------
extern "C" void kernel_launch(void* const* d_in, const int* in_sizes, int n_in,
                              void* d_out, int out_size, void* d_ws, size_t ws_size,
                              hipStream_t stream) {
    const float* x      = (const float*)d_in[0];
    const float* W_qkv  = (const float*)d_in[1];
    const float* b_qkv  = (const float*)d_in[2];
    const float* W_proj = (const float*)d_in[3];
    const float* b_proj = (const float*)d_in[4];
    float* out = (float*)d_out;

    // workspace layout (all f16): ~67.7 MB total
    _Float16* xh     = (_Float16*)d_ws;
    _Float16* wqkvT  = xh     + (size_t)NX;
    _Float16* wprojT = wqkvT  + (size_t)NWQ;
    _Float16* qb     = wprojT + (size_t)NWP;
    _Float16* kbuf   = qb     + (size_t)NX;
    _Float16* vbuf   = kbuf   + (size_t)NX;
    _Float16* yh     = vbuf   + (size_t)NX;

    {
        const size_t total = (size_t)NX + NWQ + NWP;
        dim3 grid((unsigned)((total + 255) / 256));
        k_convert<<<grid, 256, 0, stream>>>(x, W_qkv, W_proj, xh, wqkvT, wprojT);
    }
    // QKV GEMM: 128 M-blocks x 18 N-blocks, 64x128 per block
    k_qkv<<<dim3(128, 18), 256, 0, stream>>>(xh, wqkvT, b_qkv, qb, kbuf, vbuf);
    // attention: 6144 query tiles, 4 waves/block
    k_attn<<<1536, 128, 0, stream>>>(qb, kbuf, vbuf, yh);
    // projection GEMM: 128 x 6 blocks
    k_proj<<<dim3(128, 6), 256, 0, stream>>>(yh, wprojT, b_proj, out);
}
